// LeNet_12171937316916
// MI455X (gfx1250) — compile-verified, matrix operand order checked by reading, not run on hardware
//
#include <hip/hip_runtime.h>
#include <hip/hip_bf16.h>

typedef __attribute__((ext_vector_type(16))) _Float16 v16h;
typedef __attribute__((ext_vector_type(8)))  _Float16 v8h;
typedef __attribute__((ext_vector_type(8)))  float    v8f;

// ---------------- workspace layout (all 256B aligned) ----------------
// slots: 0=x 1=w1 2=w2 3=w3 4=w4 5=w5 6=h1 7=h2 8=h3 9=h4
static constexpr size_t SLOTS_OFF = 0;                                   // 16 f32
static constexpr size_t H1_OFF  = 256;                                   // 4096*6*14*14 f32
static constexpr size_t H2_OFF  = H1_OFF  + (size_t)4816896 * 4;         // 4096*16*5*5 f32
static constexpr size_t H2Q_OFF = H2_OFF  + (size_t)1638400 * 4;         // 4096*416 f16
static constexpr size_t W3Q_OFF = H2Q_OFF + (size_t)4096 * 416 * 2;      // 128*416 f16
static constexpr size_t H3_OFF  = W3Q_OFF + (size_t)128 * 416 * 2;       // 4096*120 f32
static constexpr size_t H3Q_OFF = H3_OFF  + (size_t)4096 * 120 * 4;      // 4096*128 f16
static constexpr size_t W4Q_OFF = H3Q_OFF + (size_t)4096 * 128 * 2;      // 96*128 f16
static constexpr size_t H4_OFF  = W4Q_OFF + (size_t)96 * 128 * 2;        // 4096*84 f32
static constexpr size_t H4Q_OFF = H4_OFF  + (size_t)4096 * 84 * 4;       // 4096*96 f16
static constexpr size_t W5Q_OFF = H4Q_OFF + (size_t)4096 * 96 * 2;       // 16*96 f16

__device__ __forceinline__ float slot_scale(const float* slot) {
  float m = *slot;                       // max|v| >= 0
  return m > 0.0f ? m * (1.0f / 255.0f) : 1.0f;   // QMAX = 2^(9-1)-1 = 255
}

// ---------------- init + absmax ----------------
__global__ void init_slots_kernel(float* slots) {
  if (threadIdx.x < 16) slots[threadIdx.x] = 0.0f;
}

__global__ void absmax_kernel(const float* __restrict__ src, int n, float* slot) {
  __shared__ float red[256];
  int tid = threadIdx.x;
  float m = 0.0f;
  for (int i = blockIdx.x * blockDim.x + tid; i < n; i += gridDim.x * blockDim.x)
    m = fmaxf(m, fabsf(src[i]));
  red[tid] = m;
  __syncthreads();
  for (int s = 128; s > 0; s >>= 1) {
    if (tid < s) red[tid] = fmaxf(red[tid], red[tid + s]);
    __syncthreads();
  }
  // abs values are non-negative: IEEE order == unsigned-int order
  if (tid == 0) atomicMax((unsigned int*)slot, __float_as_uint(red[0]));
}

// ---------------- conv1: quant -> conv 5x5 -> relu -> maxpool2 ----------------
// x[4096,3,32,32] * w1[6,3,5,5] -> h1[4096,6,14,14]
__global__ void conv1_pool_kernel(const float* __restrict__ x,
                                  const float* __restrict__ w1,
                                  float* __restrict__ h1,
                                  const float* __restrict__ slots) {
  int idx = blockIdx.x * blockDim.x + threadIdx.x;
  const int total = 4096 * 6 * 14 * 14;
  if (idx >= total) return;
  int px = idx % 14; int t = idx / 14;
  int py = t % 14;   t /= 14;
  int oc = t % 6;    int b = t / 6;

  float sx = slot_scale(slots + 0), sw = slot_scale(slots + 1);
  float rsx = 1.0f / sx, rsw = 1.0f / sw;

  const float* xb = x + (size_t)b * 3 * 32 * 32;
  const float* wb = w1 + oc * 75;
  int iy0 = 2 * py, ix0 = 2 * px;
  float a00 = 0.f, a01 = 0.f, a10 = 0.f, a11 = 0.f;

  for (int c = 0; c < 3; ++c) {
    float p[6][6];
    #pragma unroll
    for (int dy = 0; dy < 6; ++dy)
      #pragma unroll
      for (int dx = 0; dx < 6; ++dx)
        p[dy][dx] = rintf(xb[c * 1024 + (iy0 + dy) * 32 + (ix0 + dx)] * rsx) * sx;
    #pragma unroll
    for (int ky = 0; ky < 5; ++ky)
      #pragma unroll
      for (int kx = 0; kx < 5; ++kx) {
        float wq = rintf(wb[c * 25 + ky * 5 + kx] * rsw) * sw;
        a00 = fmaf(p[ky][kx],         wq, a00);
        a01 = fmaf(p[ky][kx + 1],     wq, a01);
        a10 = fmaf(p[ky + 1][kx],     wq, a10);
        a11 = fmaf(p[ky + 1][kx + 1], wq, a11);
      }
  }
  float m = fmaxf(fmaxf(a00, a01), fmaxf(a10, a11));
  h1[idx] = fmaxf(m, 0.0f);  // relu(maxpool) == maxpool(relu)
}

// ---------------- conv2: h1[4096,6,14,14] * w2[16,6,5,5] -> h2[4096,16,5,5] ----------------
__global__ void conv2_pool_kernel(const float* __restrict__ h1,
                                  const float* __restrict__ w2,
                                  float* __restrict__ h2,
                                  const float* __restrict__ slots) {
  int idx = blockIdx.x * blockDim.x + threadIdx.x;
  const int total = 4096 * 16 * 5 * 5;
  if (idx >= total) return;
  int px = idx % 5; int t = idx / 5;
  int py = t % 5;   t /= 5;
  int oc = t % 16;  int b = t / 16;

  float sx = slot_scale(slots + 6), sw = slot_scale(slots + 2);
  float rsx = 1.0f / sx, rsw = 1.0f / sw;

  const float* xb = h1 + (size_t)b * 6 * 196;
  const float* wb = w2 + oc * 150;
  int iy0 = 2 * py, ix0 = 2 * px;
  float a00 = 0.f, a01 = 0.f, a10 = 0.f, a11 = 0.f;

  for (int c = 0; c < 6; ++c) {
    float p[6][6];
    #pragma unroll
    for (int dy = 0; dy < 6; ++dy)
      #pragma unroll
      for (int dx = 0; dx < 6; ++dx)
        p[dy][dx] = rintf(xb[c * 196 + (iy0 + dy) * 14 + (ix0 + dx)] * rsx) * sx;
    #pragma unroll
    for (int ky = 0; ky < 5; ++ky)
      #pragma unroll
      for (int kx = 0; kx < 5; ++kx) {
        float wq = rintf(wb[c * 25 + ky * 5 + kx] * rsw) * sw;
        a00 = fmaf(p[ky][kx],         wq, a00);
        a01 = fmaf(p[ky][kx + 1],     wq, a01);
        a10 = fmaf(p[ky + 1][kx],     wq, a10);
        a11 = fmaf(p[ky + 1][kx + 1], wq, a11);
      }
  }
  float m = fmaxf(fmaxf(a00, a01), fmaxf(a10, a11));
  h2[idx] = fmaxf(m, 0.0f);
}

// ---------------- quantize f32 -> f16 integer representatives, zero-padded ----------------
__global__ void quantize_f16_kernel(const float* __restrict__ src, _Float16* __restrict__ dst,
                                    int R, int C, int Rpad, int Cpad,
                                    const float* __restrict__ maxslot) {
  int i = blockIdx.x * blockDim.x + threadIdx.x;
  int total = Rpad * Cpad;
  if (i >= total) return;
  int r = i / Cpad, c = i % Cpad;
  float v = 0.0f;
  if (r < R && c < C) {
    float s = slot_scale(maxslot);
    v = rintf(src[(size_t)r * C + c] / s);   // integer in [-255,255], exact in f16
  }
  dst[i] = (_Float16)v;
}

// ---------------- WMMA GEMM: C[m,n] = (sum_k Aint[m,k]*Bint[n,k]) * sA*sB ----------------
// A: [Mtiles*16][ldA] f16 (row-major), B: [Ntiles*16][ldB] f16 (row-major, = W[n][k]).
// One wave computes one 16x16 tile. Fragment layout per CDNA5 ISA §7.12.2:
//   lane = (row&15) | (khalf<<4); elems 0..7 -> K=khalf*8+0..7, elems 8..15 -> K=16+khalf*8+0..7
__device__ __forceinline__ v16h load_frag(const _Float16* __restrict__ base,
                                          int ld, int row, int kh, int k0) {
  const _Float16* p = base + (size_t)row * ld + k0 + kh;   // 16B aligned: ld % 8 == 0
  v8h lo = *reinterpret_cast<const v8h*>(p);
  v8h hi = *reinterpret_cast<const v8h*>(p + 16);
  return __builtin_shufflevector(lo, hi, 0, 1, 2, 3, 4, 5, 6, 7,
                                         8, 9, 10, 11, 12, 13, 14, 15);
}

__global__ void gemm_wmma_kernel(const _Float16* __restrict__ A,
                                 const _Float16* __restrict__ B,
                                 float* __restrict__ C,
                                 int Ntiles, int Kpad, int ldA, int ldB,
                                 int Nout, int ldC,
                                 const float* __restrict__ maxA,
                                 const float* __restrict__ maxB, int relu) {
  int wave = (blockIdx.x * blockDim.x + threadIdx.x) >> 5;
  int lane = threadIdx.x & 31;
  int mt = wave / Ntiles;
  int nt = wave % Ntiles;
  const _Float16* Ap = A + (size_t)mt * 16 * ldA;
  const _Float16* Bp = B + (size_t)nt * 16 * ldB;
  int row = lane & 15;
  int kh  = (lane >> 4) * 8;

  v8f acc = {0.f, 0.f, 0.f, 0.f, 0.f, 0.f, 0.f, 0.f};
  for (int k0 = 0; k0 < Kpad; k0 += 32) {
    v16h a = load_frag(Ap, ldA, row, kh, k0);
    v16h b = load_frag(Bp, ldB, row, kh, k0);
    acc = __builtin_amdgcn_wmma_f32_16x16x32_f16(false, a, false, b,
                                                 (short)0, acc, false, false);
  }

  float sc = slot_scale(maxA) * slot_scale(maxB);
  int n  = nt * 16 + row;          // C: lane low nibble = column
  int m0 = mt * 16 + kh;           // C: vgpr j -> row m0+j (kh = 0 or 8)
  if (n < Nout) {
    #pragma unroll
    for (int j = 0; j < 8; ++j) {
      float v = acc[j] * sc;
      if (relu) v = fmaxf(v, 0.0f);
      C[(size_t)(m0 + j) * ldC + n] = v;
    }
  }
}

// ---------------- host launch ----------------
extern "C" void kernel_launch(void* const* d_in, const int* in_sizes, int n_in,
                              void* d_out, int out_size, void* d_ws, size_t ws_size,
                              hipStream_t stream) {
  (void)in_sizes; (void)n_in; (void)out_size; (void)ws_size;
  const float* x  = (const float*)d_in[0];
  const float* w1 = (const float*)d_in[1];
  const float* w2 = (const float*)d_in[2];
  const float* w3 = (const float*)d_in[3];
  const float* w4 = (const float*)d_in[4];
  const float* w5 = (const float*)d_in[5];
  float* out = (float*)d_out;

  char* ws = (char*)d_ws;
  float*    slots = (float*)(ws + SLOTS_OFF);
  float*    h1  = (float*)(ws + H1_OFF);
  float*    h2  = (float*)(ws + H2_OFF);
  _Float16* h2q = (_Float16*)(ws + H2Q_OFF);
  _Float16* w3q = (_Float16*)(ws + W3Q_OFF);
  float*    h3  = (float*)(ws + H3_OFF);
  _Float16* h3q = (_Float16*)(ws + H3Q_OFF);
  _Float16* w4q = (_Float16*)(ws + W4Q_OFF);
  float*    h4  = (float*)(ws + H4_OFF);
  _Float16* h4q = (_Float16*)(ws + H4Q_OFF);
  _Float16* w5q = (_Float16*)(ws + W5Q_OFF);

  const int T = 256;
  auto blocks = [](int n) { return (n + 255) / 256; };
  auto rblocks = [](int n) { int b = (n + 255) / 256; return b > 2048 ? 2048 : b; };

  init_slots_kernel<<<1, 32, 0, stream>>>(slots);

  // per-tensor abs-max for fake-quant scales
  absmax_kernel<<<rblocks(4096*3*32*32), T, 0, stream>>>(x,  4096*3*32*32, slots + 0);
  absmax_kernel<<<rblocks(450),   T, 0, stream>>>(w1, 450,   slots + 1);
  absmax_kernel<<<rblocks(2400),  T, 0, stream>>>(w2, 2400,  slots + 2);
  absmax_kernel<<<rblocks(48000), T, 0, stream>>>(w3, 48000, slots + 3);
  absmax_kernel<<<rblocks(10080), T, 0, stream>>>(w4, 10080, slots + 4);
  absmax_kernel<<<rblocks(840),   T, 0, stream>>>(w5, 840,   slots + 5);

  // stage 1: conv1 + relu + pool
  conv1_pool_kernel<<<blocks(4096*6*14*14), T, 0, stream>>>(x, w1, h1, slots);
  absmax_kernel<<<rblocks(4096*6*14*14), T, 0, stream>>>(h1, 4096*6*14*14, slots + 6);

  // stage 2: conv2 + relu + pool
  conv2_pool_kernel<<<blocks(4096*16*25), T, 0, stream>>>(h1, w2, h2, slots);
  absmax_kernel<<<rblocks(4096*16*25), T, 0, stream>>>(h2, 4096*16*25, slots + 7);

  // stage 3: conv3 as GEMM [4096,400]x[120,400]^T on WMMA (pad K 400->416, N 120->128)
  quantize_f16_kernel<<<blocks(4096*416), T, 0, stream>>>(h2, h2q, 4096, 400, 4096, 416, slots + 7);
  quantize_f16_kernel<<<blocks(128*416),  T, 0, stream>>>(w3, w3q, 120, 400, 128, 416, slots + 3);
  gemm_wmma_kernel<<<(256*8)/8, T, 0, stream>>>(h2q, w3q, h3,
      /*Ntiles=*/8, /*Kpad=*/416, /*ldA=*/416, /*ldB=*/416,
      /*Nout=*/120, /*ldC=*/120, slots + 7, slots + 3, /*relu=*/1);
  absmax_kernel<<<rblocks(4096*120), T, 0, stream>>>(h3, 4096*120, slots + 8);

  // stage 4: fc4 [4096,120]x[84,120]^T (pad K 120->128, N 84->96)
  quantize_f16_kernel<<<blocks(4096*128), T, 0, stream>>>(h3, h3q, 4096, 120, 4096, 128, slots + 8);
  quantize_f16_kernel<<<blocks(96*128),   T, 0, stream>>>(w4, w4q, 84, 120, 96, 128, slots + 4);
  gemm_wmma_kernel<<<(256*6)/8, T, 0, stream>>>(h3q, w4q, h4,
      /*Ntiles=*/6, /*Kpad=*/128, /*ldA=*/128, /*ldB=*/128,
      /*Nout=*/84, /*ldC=*/84, slots + 8, slots + 4, /*relu=*/1);
  absmax_kernel<<<rblocks(4096*84), T, 0, stream>>>(h4, 4096*84, slots + 9);

  // stage 5: fc5 [4096,84]x[10,84]^T (pad K 84->96, N 10->16) -> logits
  quantize_f16_kernel<<<blocks(4096*96), T, 0, stream>>>(h4, h4q, 4096, 84, 4096, 96, slots + 9);
  quantize_f16_kernel<<<blocks(16*96),   T, 0, stream>>>(w5, w5q, 10, 84, 16, 96, slots + 5);
  gemm_wmma_kernel<<<(256*1)/8, T, 0, stream>>>(h4q, w5q, out,
      /*Ntiles=*/1, /*Kpad=*/96, /*ldA=*/96, /*ldB=*/96,
      /*Nout=*/10, /*ldC=*/10, slots + 9, slots + 5, /*relu=*/0);
}
